// NeighborhoodAttention2D_19636590477337
// MI455X (gfx1250) — matrix operand order repaired and checked by reading, not compiled, not run on hardware
//
#include <hip/hip_runtime.h>
#include <hip/hip_bf16.h>

// ---------------------------------------------------------------------------
// NATTEN-2D (K=7, D=2) on gfx1250, all matrix work via v_wmma_f32_16x16x32_f16.
// Pipeline: cvt_x -> cvt_w -> qkv_gemm (WMMA) -> natten (fused WMMA attention).
// K-union gather uses GLOBAL_LOAD_ASYNC_TO_LDS_B128 (ASYNCcnt-tracked).
// ---------------------------------------------------------------------------

typedef __attribute__((ext_vector_type(16))) _Float16 v16h;
typedef __attribute__((ext_vector_type(8)))  float    v8f;

union V16 { v16h h; uint4 u[2]; };

#define NB    2
#define CIN   128
#define COUT  64
#define HD    112
#define WD    112
#define HW    (HD*WD)        // 12544
#define NPIX  (NB*HW)        // 25088
#define MTILES (NPIX/16)     // 1568

#define NCW  28              // union key-column width per 16-query strip
#define NKEY 196             // 7 rows * 28 cols
#define NKP  224             // padded to 7 WMMA k-steps of 32

#ifndef __has_builtin
#define __has_builtin(x) 0
#endif
#if __has_builtin(__builtin_amdgcn_global_load_async_to_lds_b128)
#define HAVE_ASYNC_LDS 1
#else
#define HAVE_ASYNC_LDS 0
#endif
#if __has_builtin(__builtin_amdgcn_s_wait_asynccnt)
#define WAIT_ASYNC() __builtin_amdgcn_s_wait_asynccnt(0)
#elif HAVE_ASYNC_LDS
#define WAIT_ASYNC() asm volatile("s_wait_asynccnt 0x0" ::: "memory")
#else
#define WAIT_ASYNC()
#endif

#if HAVE_ASYNC_LDS
// Builtin prototype (from hipcc diagnostic): arg0 is v4i in addrspace(1).
typedef int async_v4i __attribute__((vector_size(16)));
typedef __attribute__((address_space(1))) async_v4i* g_v4i_p;
typedef __attribute__((address_space(3))) async_v4i* l_v4i_p;
#endif

// window_start for L=112, K=7, D=2  (nh*D = 6; a=112, b=0 -> right = 98+parity)
__device__ __forceinline__ int wstart(int idx) {
  int ni = idx - 6;
  int par = idx & 1;
  if (ni < 0) return par;
  if (idx + 6 >= WD) return 98 + par;   // HD == WD == 112
  return ni;
}

// -------- kernel A1: x (B,Cin,H,W) f32 -> xh [pixel][Cin] f16 (transposed) --
__global__ void cvt_x(const float* __restrict__ x, _Float16* __restrict__ xh) {
  int i = blockIdx.x * 256 + threadIdx.x;          // exact grid
  int p = i % HW;
  int c = (i / HW) % CIN;
  int b = i / (HW * CIN);
  xh[(b * HW + p) * CIN + c] = (_Float16)x[i];
}

// -------- kernel A2: Wq|Wk|Wv (64x128 f32 each) -> wh [192][128] f16 --------
__global__ void cvt_w(const float* __restrict__ Wq, const float* __restrict__ Wk,
                      const float* __restrict__ Wv, _Float16* __restrict__ wh) {
  int i = blockIdx.x * 256 + threadIdx.x;          // 0..24575 exact
  int n = i / CIN, c = i % CIN;
  float v = (n < 64) ? Wq[n * CIN + c]
          : (n < 128) ? Wk[(n - 64) * CIN + c]
          : Wv[(n - 128) * CIN + c];
  wh[i] = (_Float16)v;
}

// -------- kernel B: QKV projection GEMM, 16-pixel tile per wave -------------
// out[m, n] = sum_c xh[m,c] * wh[n,c];  q gets (val+bias)*0.125, k/v get +bias
__global__ __launch_bounds__(128) void qkv_gemm(
    const _Float16* __restrict__ xh, const _Float16* __restrict__ wh,
    const float* __restrict__ bq, const float* __restrict__ bk,
    const float* __restrict__ bv,
    _Float16* __restrict__ qh, _Float16* __restrict__ kh,
    _Float16* __restrict__ vh) {
  int lane = threadIdx.x & 31;
  int wid  = threadIdx.x >> 5;
  int mtile = blockIdx.x * 4 + wid;
  int half = lane >> 4, l16 = lane & 15;

  // A fragments: 16x32 f16; lane holds row l16, halves 0..7 -> K = ks*32+8*half..,
  // halves 8..15 -> +16.
  V16 a[4];
  const _Float16* ap = xh + (size_t)(mtile * 16 + l16) * CIN;
#pragma unroll
  for (int ks = 0; ks < 4; ++ks) {
    int ch0 = ks * 32 + 8 * half;
    a[ks].u[0] = *(const uint4*)(ap + ch0);
    a[ks].u[1] = *(const uint4*)(ap + ch0 + 16);
  }

#pragma unroll
  for (int nt = 0; nt < 12; ++nt) {
    v8f acc = {};
    int ncol = nt * 16 + l16;                       // global output column
    const _Float16* bp = wh + ncol * CIN + 8 * half;
#pragma unroll
    for (int ks = 0; ks < 4; ++ks) {
      V16 bf;                                       // B 32x16: lane=col, halves=K
      bf.u[0] = *(const uint4*)(bp + ks * 32);
      bf.u[1] = *(const uint4*)(bp + ks * 32 + 16);
      acc = __builtin_amdgcn_wmma_f32_16x16x32_f16(
          false, a[ks].h, false, bf.h, (short)0, acc, false, false);
    }
    int qsel = nt >> 2;
    int cc = (nt & 3) * 16 + l16;
    float bias = (qsel == 0) ? bq[cc] : (qsel == 1) ? bk[cc] : bv[cc];
    float scale = (qsel == 0) ? 0.125f : 1.0f;      // 64^-0.5 folded into q
    _Float16* dst = (qsel == 0) ? qh : (qsel == 1) ? kh : vh;
#pragma unroll
    for (int r = 0; r < 8; ++r) {                   // C/D: row = r + 8*half
      int pix = mtile * 16 + r + 8 * half;
      dst[(size_t)pix * COUT + cc] = (_Float16)((acc[r] + bias) * scale);
    }
  }
}

// -------- kernel C: fused neighborhood attention per 16-query strip ---------
__global__ __launch_bounds__(32) void natten(
    const _Float16* __restrict__ qh, const _Float16* __restrict__ kh,
    const _Float16* __restrict__ vh, float* __restrict__ out) {
  __shared__ __align__(16) _Float16 Ku[NKP][COUT];   // gathered K  [key][ch]
  __shared__ __align__(16) _Float16 Vt[COUT][NKP];   // gathered V^T[ch][key]
  __shared__ __align__(16) _Float16 Pl[16][NKP];     // softmax probs [row][key]

  int lane = threadIdx.x;
  int w0 = blockIdx.x * 16;
  int h  = blockIdx.y;
  int b  = blockIdx.z;
  int half = lane >> 4, l16 = lane & 15;

  int ihs = wstart(h);                               // shared key rows
  int cmin = 1 << 30;
#pragma unroll
  for (int w = 0; w < 16; ++w) cmin = min(cmin, wstart(w0 + w));

  // ---- gather union keys (7 rows x 28 cols) into LDS ----
  for (int t = lane; t < NKEY; t += 32) {
    int kr = t / NCW, kc = t % NCW;
    int hk = ihs + 2 * kr;
    int wk = min(cmin + kc, WD - 1);
    size_t src = (size_t)(b * HW + hk * WD + wk) * COUT;
#if HAVE_ASYNC_LDS
    // CDNA5 async global->LDS copy (ASYNCcnt-tracked), K rows land in LDS
    // without a VGPR round trip.
#pragma unroll
    for (int j = 0; j < 8; ++j)
      __builtin_amdgcn_global_load_async_to_lds_b128(
          (g_v4i_p)(kh + src + j * 8),
          (l_v4i_p)(&Ku[t][j * 8]), 0, 0);
#else
    union { uint4 u[8]; } kb;
#pragma unroll
    for (int j = 0; j < 8; ++j) kb.u[j] = *(const uint4*)(kh + src + j * 8);
#pragma unroll
    for (int j = 0; j < 8; ++j) *(uint4*)&Ku[t][j * 8] = kb.u[j];
#endif
    // V must be transposed on the way into LDS -> synchronous gather/scatter
    union { uint4 u[8]; _Float16 hx[64]; } vb;
#pragma unroll
    for (int j = 0; j < 8; ++j) vb.u[j] = *(const uint4*)(vh + src + j * 8);
#pragma unroll
    for (int c = 0; c < COUT; ++c) Vt[c][t] = vb.hx[c];
  }
  // zero the pad keys so WMMA never multiplies NaN garbage
  for (int t = NKEY + lane; t < NKP; t += 32) {
    uint4 z; z.x = z.y = z.z = z.w = 0u;
#pragma unroll
    for (int j = 0; j < 8; ++j) *(uint4*)&Ku[t][j * 8] = z;
#pragma unroll
    for (int c = 0; c < COUT; ++c) Vt[c][t] = (_Float16)0.0f;
  }
  WAIT_ASYNC();
  __syncthreads();

  // ---- scores = Q(16x64) @ Ku^T(64x224) via WMMA ----
  V16 qa[2];
  const _Float16* qp = qh + (size_t)(b * HW + h * WD + w0 + l16) * COUT;
#pragma unroll
  for (int ks = 0; ks < 2; ++ks) {
    int ch0 = ks * 32 + 8 * half;
    qa[ks].u[0] = *(const uint4*)(qp + ch0);
    qa[ks].u[1] = *(const uint4*)(qp + ch0 + 16);
  }
  v8f s[14] = {};
#pragma unroll
  for (int ks = 0; ks < 2; ++ks) {
#pragma unroll
    for (int nt = 0; nt < 14; ++nt) {
      V16 bf;
      const _Float16* kp = &Ku[nt * 16 + l16][ks * 32 + 8 * half];
      bf.u[0] = *(const uint4*)(kp);
      bf.u[1] = *(const uint4*)(kp + 16);
      s[nt] = __builtin_amdgcn_wmma_f32_16x16x32_f16(
          false, qa[ks].h, false, bf.h, (short)0, s[nt], false, false);
    }
  }

  // ---- exact window mask + softmax (rows live in 16-lane halves) ----
  int iwq[8];
#pragma unroll
  for (int r = 0; r < 8; ++r) iwq[r] = wstart(w0 + r + 8 * half);
#pragma unroll
  for (int nt = 0; nt < 14; ++nt) {
    int kk = nt * 16 + l16;
    int ckey = cmin + (kk % NCW);
#pragma unroll
    for (int r = 0; r < 8; ++r) {
      int d = ckey - iwq[r];
      bool valid = (kk < NKEY) && (d >= 0) && (d <= 12) && ((d & 1) == 0);
      s[nt][r] = valid ? s[nt][r] : -1e30f;
    }
  }
  float rsum[8];
#pragma unroll
  for (int r = 0; r < 8; ++r) {
    float m = -1e30f;
#pragma unroll
    for (int nt = 0; nt < 14; ++nt) m = fmaxf(m, s[nt][r]);
    m = fmaxf(m, __shfl_xor(m, 1));
    m = fmaxf(m, __shfl_xor(m, 2));
    m = fmaxf(m, __shfl_xor(m, 4));
    m = fmaxf(m, __shfl_xor(m, 8));
    float sm = 0.f;
#pragma unroll
    for (int nt = 0; nt < 14; ++nt) {
      float p = expf(s[nt][r] - m);                  // invalid -> exp(-huge)=0
      s[nt][r] = p;
      sm += p;
    }
    sm += __shfl_xor(sm, 1);
    sm += __shfl_xor(sm, 2);
    sm += __shfl_xor(sm, 4);
    sm += __shfl_xor(sm, 8);
    rsum[r] = sm;                                    // >= 1, safe divisor
  }
  // C-layout -> A-layout reshuffle through LDS
#pragma unroll
  for (int nt = 0; nt < 14; ++nt) {
    int kk = nt * 16 + l16;
#pragma unroll
    for (int r = 0; r < 8; ++r) Pl[r + 8 * half][kk] = (_Float16)s[nt][r];
  }
  __syncthreads();

  // ---- out = P(16x224) @ V(224x64) via WMMA, normalize at store ----
  V16 pa[7];
#pragma unroll
  for (int ks = 0; ks < 7; ++ks) {
    const _Float16* pp = &Pl[l16][ks * 32 + 8 * half];
    pa[ks].u[0] = *(const uint4*)(pp);
    pa[ks].u[1] = *(const uint4*)(pp + 16);
  }
#pragma unroll
  for (int nt2 = 0; nt2 < 4; ++nt2) {
    v8f o = {};
#pragma unroll
    for (int ks = 0; ks < 7; ++ks) {
      V16 bf;
      const _Float16* vp = &Vt[nt2 * 16 + l16][ks * 32 + 8 * half];
      bf.u[0] = *(const uint4*)(vp);
      bf.u[1] = *(const uint4*)(vp + 16);
      o = __builtin_amdgcn_wmma_f32_16x16x32_f16(
          false, pa[ks].h, false, bf.h, (short)0, o, false, false);
    }
    int ch = nt2 * 16 + l16;
#pragma unroll
    for (int r = 0; r < 8; ++r) {
      int w = w0 + r + 8 * half;
      out[(size_t)((b * COUT + ch) * HD + h) * WD + w] = o[r] / rsum[r];
    }
  }
}

// ---------------------------------------------------------------------------
extern "C" void kernel_launch(void* const* d_in, const int* in_sizes, int n_in,
                              void* d_out, int out_size, void* d_ws, size_t ws_size,
                              hipStream_t stream) {
  const float* x  = (const float*)d_in[0];
  const float* Wq = (const float*)d_in[1];
  const float* bq = (const float*)d_in[2];
  const float* Wk = (const float*)d_in[3];
  const float* bk = (const float*)d_in[4];
  const float* Wv = (const float*)d_in[5];
  const float* bv = (const float*)d_in[6];
  float* out = (float*)d_out;

  char* ws = (char*)d_ws;
  // layout (all 256B-aligned): xh 6.4MB | wh 48KB | qh/kh/vh 3.2MB each (~16MB)
  _Float16* xh = (_Float16*)ws;
  _Float16* wh = (_Float16*)(ws + (size_t)NPIX * CIN * 2);
  _Float16* qh = (_Float16*)(ws + (size_t)NPIX * CIN * 2 + 192 * CIN * 2);
  _Float16* kh = qh + (size_t)NPIX * COUT;
  _Float16* vh = kh + (size_t)NPIX * COUT;

  cvt_x<<<(NB * CIN * HW) / 256, 256, 0, stream>>>(x, xh);
  cvt_w<<<(192 * CIN) / 256, 256, 0, stream>>>(Wq, Wk, Wv, wh);
  qkv_gemm<<<MTILES / 4, 128, 0, stream>>>(xh, wh, bq, bk, bv, qh, kh, vh);
  dim3 g(WD / 16, HD, NB);
  natten<<<g, 32, 0, stream>>>(qh, kh, vh, out);
}
